// GAT_TimeSeriesLayer_11613591568918
// MI455X (gfx1250) — matrix operand order repaired and checked by reading, not compiled, not run on hardware
//
#include <hip/hip_runtime.h>
#include <hip/hip_bf16.h>
#include <cstddef>
#include <cstdint>

// ---------------- problem dims ----------------
#define S_OBS 8
#define S_PRED 12
#define NN 2048

typedef __attribute__((ext_vector_type(16))) _Float16 v16h;
typedef __attribute__((ext_vector_type(8)))  float    v8f;

// ---------------- workspace layout (float elements) ----------------
static constexpr size_t OFF_H1  = 0;                            // [8][2048][16]
static constexpr size_t OFF_G1  = OFF_H1  + (size_t)8*2048*16;  // [8][2048][30]
static constexpr size_t OFF_X1  = OFF_G1  + (size_t)8*2048*30;  // gat out  [8][2048][30]
static constexpr size_t OFF_X3  = OFF_X1  + (size_t)8*2048*30;  // mlp out  [8][2048][30]
static constexpr size_t OFF_G2  = OFF_X3  + (size_t)8*2048*30;  // gru2 out [8][2048][30]
static constexpr size_t OFF_X6  = OFF_G2  + (size_t)8*2048*30;  // conv out [12][2048][30]
static constexpr size_t OFF_AS  = OFF_X6  + (size_t)12*2048*30; // a_src [8][2048]
static constexpr size_t OFF_AD  = OFF_AS  + (size_t)8*2048;     // a_dst [8][2048]
static constexpr size_t OFF_XWH = OFF_AD  + (size_t)8*2048;     // f16 xw TRANSPOSED [8][32][2048] (halfs)

// ---------------- fc1: x[s,n,2] -> relu(W x + b) [s,n,16] ----------------
__global__ void fc1_kernel(const float* __restrict__ x, const float* __restrict__ w,
                           const float* __restrict__ b, float* __restrict__ h1) {
  int t = blockIdx.x * blockDim.x + threadIdx.x;   // t = s*2048 + n, 16384 threads
  float x0 = x[(size_t)t * 2 + 0];
  float x1 = x[(size_t)t * 2 + 1];
#pragma unroll
  for (int o = 0; o < 16; ++o) {
    float v = b[o] + w[o * 2 + 0] * x0 + w[o * 2 + 1] * x1;
    h1[(size_t)t * 16 + o] = v > 0.f ? v : 0.f;
  }
}

// ---------------- GRU (torch gate order r,z,n), thread-per-node, recurrence in regs ----------------
template <int DIN>
__global__ void gru_kernel(const float* __restrict__ in,   // [8][2048][DIN]
                           const float* __restrict__ wih,  // [90][DIN]
                           const float* __restrict__ whh,  // [90][30]
                           const float* __restrict__ bih, const float* __restrict__ bhh,
                           float* __restrict__ out) {      // [8][2048][30] = relu(h_s)
  __shared__ float s_wih[90 * DIN];
  __shared__ float s_whh[90 * 30];
  __shared__ float s_bih[90], s_bhh[90];
  for (int i = threadIdx.x; i < 90 * DIN; i += blockDim.x) s_wih[i] = wih[i];
  for (int i = threadIdx.x; i < 90 * 30; i += blockDim.x) s_whh[i] = whh[i];
  for (int i = threadIdx.x; i < 90; i += blockDim.x) { s_bih[i] = bih[i]; s_bhh[i] = bhh[i]; }
  __syncthreads();
  int n = blockIdx.x * blockDim.x + threadIdx.x;   // 0..2047
  float h[30];
#pragma unroll
  for (int k = 0; k < 30; ++k) h[k] = 0.f;
  for (int s = 0; s < S_OBS; ++s) {
    float xv[DIN];
#pragma unroll
    for (int d = 0; d < DIN; ++d) xv[d] = in[((size_t)s * NN + n) * DIN + d];
    float hnew[30];
#pragma unroll
    for (int k = 0; k < 30; ++k) {
      float ir = s_bih[k], iz = s_bih[30 + k], inn = s_bih[60 + k];
#pragma unroll
      for (int d = 0; d < DIN; ++d) {
        ir  += s_wih[k * DIN + d]        * xv[d];
        iz  += s_wih[(30 + k) * DIN + d] * xv[d];
        inn += s_wih[(60 + k) * DIN + d] * xv[d];
      }
      float hr = s_bhh[k], hz = s_bhh[30 + k], hn = s_bhh[60 + k];
#pragma unroll
      for (int d = 0; d < 30; ++d) {
        hr += s_whh[k * 30 + d]        * h[d];
        hz += s_whh[(30 + k) * 30 + d] * h[d];
        hn += s_whh[(60 + k) * 30 + d] * h[d];
      }
      float r  = 1.f / (1.f + __expf(-(ir + hr)));
      float z  = 1.f / (1.f + __expf(-(iz + hz)));
      float nn2 = tanhf(inn + r * hn);
      hnew[k] = (1.f - z) * nn2 + z * h[k];
    }
#pragma unroll
    for (int k = 0; k < 30; ++k) {
      h[k] = hnew[k];
      out[((size_t)s * NN + n) * 30 + k] = h[k] > 0.f ? h[k] : 0.f;
    }
  }
}

// ---------------- GAT prep: xw = g1 @ W^T, attention logits, TRANSPOSED f16 xw ----------------
__global__ void gat_pre_kernel(const float* __restrict__ g1, const float* __restrict__ gat_w,
                               const float* __restrict__ att_s, const float* __restrict__ att_d,
                               float* __restrict__ a_s, float* __restrict__ a_d,
                               _Float16* __restrict__ xwh) {
  __shared__ float sw[900], ssrc[30], sdst[30];
  for (int i = threadIdx.x; i < 900; i += blockDim.x) sw[i] = gat_w[i];
  if (threadIdx.x < 30) { ssrc[threadIdx.x] = att_s[threadIdx.x]; sdst[threadIdx.x] = att_d[threadIdx.x]; }
  __syncthreads();
  int t = blockIdx.x * blockDim.x + threadIdx.x;   // s*2048 + n
  int s = t >> 11;
  int n = t & (NN - 1);
  float g[30];
#pragma unroll
  for (int d = 0; d < 30; ++d) g[d] = g1[(size_t)t * 30 + d];
  float as = 0.f, ad = 0.f;
  _Float16* xq = xwh + (size_t)s * 32 * NN + n;    // column n of [32][2048]
#pragma unroll
  for (int h = 0; h < 30; ++h) {
    float v = 0.f;
#pragma unroll
    for (int d = 0; d < 30; ++d) v += g[d] * sw[h * 30 + d];
    xq[(size_t)h * NN] = (_Float16)v;
    as += v * ssrc[h];
    ad += v * sdst[h];
  }
  xq[(size_t)30 * NN] = (_Float16)0.f;
  xq[(size_t)31 * NN] = (_Float16)0.f;
  a_s[t] = as;
  a_d[t] = ad;
}

// ---------------- fused masked-softmax attention aggregation (WMMA f16 + async LDS) ----------------
// One wave per (s, 16-target tile). Adjacency tile (32 src rows x 16 tgt cols) is streamed with
// double-buffered global_load_async_to_lds_b128; exp is branch-free; B comes as 2x b128 per lane
// from the transposed f16 xw. x1[s,i,d] = (A@B)[m,d]/rowsum(A) + bias.
__global__ void gat_attn_wmma_kernel(const int* __restrict__ adj,
                                     const float* __restrict__ a_s, const float* __restrict__ a_d,
                                     const _Float16* __restrict__ xwh,
                                     const float* __restrict__ gat_bias,
                                     float* __restrict__ x1) {
  constexpr int WAVES = 4;
  __shared__ int   s_adj[WAVES][2][32][16];   // double-buffered per-wave adj tiles (16 KB)
  __shared__ float s_as[NN];                  // a_src row for this s (8 KB)
  __shared__ float sden[WAVES][16];

  int wave = threadIdx.x >> 5;
  int lane = threadIdx.x & 31;
  int tile = blockIdx.x * WAVES + wave;        // 0..1023; block => same s (4 | 128)
  int s  = tile >> 7;
  int i0 = (tile & 127) << 4;
  int hs = lane >> 4;
  int lr = lane & 15;
  int i_row = i0 + lr;                         // A-row m == target i

  // stage a_src[s][0:2048] into LDS (whole block, same s)
  for (int i = threadIdx.x; i < NN; i += blockDim.x) s_as[i] = a_s[(size_t)s * NN + i];
  __syncthreads();

  float ad = a_d[(size_t)s * NN + i_row];
  const int*      adjS = adj + (size_t)s * NN * NN;
  const _Float16* xwT  = xwh + (size_t)s * 32 * NN;

  // async issue of one 32x16 int tile into buffer `buf` (4 x b128 per wave, coalesced 16B/lane)
  auto issue_tile = [&](int jc, int buf) {
#pragma unroll
    for (int q = 0; q < 4; ++q) {
      int tidx = q * 32 + lane;      // 0..127
      int row  = tidx >> 2;          // 0..31
      int seg  = tidx & 3;           // 16B segment
      const int* gp = adjS + (size_t)(jc + row) * NN + i0 + seg * 4;
      unsigned  lp  = (unsigned)(uintptr_t)(&s_adj[wave][buf][row][seg * 4]);
      asm volatile("global_load_async_to_lds_b128 %0, %1, off"
                   :: "v"(lp), "v"(gp) : "memory");
    }
  };

  v8f c0 = {}; v8f c1 = {};
  float den = 0.f;

  issue_tile(0, 0);
  for (int j0 = 0; j0 < NN; j0 += 32) {
    int buf = (j0 >> 5) & 1;
    if (j0 + 32 < NN) {
      issue_tile(j0 + 32, buf ^ 1);                    // prefetch next chunk
      asm volatile("s_wait_asynccnt 0x4" ::: "memory"); // current chunk's 4 are done
    } else {
      asm volatile("s_wait_asynccnt 0x0" ::: "memory");
    }
    // A fragment: contiguous k = 16*hs + s (matches B); branch-free masked exp
    v16h a;
#pragma unroll
    for (int g = 0; g < 8; ++g) {
      int kb = 2 * g + hs * 16;
#pragma unroll
      for (int e = 0; e < 2; ++e) {
        int k = kb + e;
        int j = j0 + k;
        int vv = s_adj[wave][buf][k][lr];
        float msk = (float)((vv != 0) | (j == i_row));   // incl. self-loop
        float sc = s_as[j] + ad;
        sc = sc > 0.f ? sc : 0.2f * sc;                  // leaky_relu
        float p = __expf(sc) * msk;                      // exp always executed
        den += p;
        a[2 * g + e] = (_Float16)p;
      }
    }
    // B fragments from transposed xw: 16 contiguous halfs per lane (32B aligned)
    v16h b0 = *(const v16h*)(xwT + (size_t)lr        * NN + j0 + hs * 16);
    v16h b1 = *(const v16h*)(xwT + (size_t)(16 + lr) * NN + j0 + hs * 16);
    c0 = __builtin_amdgcn_wmma_f32_16x16x32_f16(false, a, false, b0, (short)0, c0, false, false);
    c1 = __builtin_amdgcn_wmma_f32_16x16x32_f16(false, a, false, b1, (short)0, c1, false, false);
  }
  den += __shfl_xor(den, 16, 32);              // combine K-halves per target row
  if (lane < 16) sden[wave][lr] = den;
  __syncthreads();
#pragma unroll
  for (int r = 0; r < 8; ++r) {
    int Mrow = r + 8 * hs;                     // C/D layout: VGPR r -> M=r (lo lanes) / r+8 (hi)
    int i = i0 + Mrow;
    float inv = 1.f / sden[wave][Mrow];
    int d0 = lr;
    x1[((size_t)s * NN + i) * 30 + d0] = c0[r] * inv + gat_bias[d0];
    int d1 = lr + 16;
    if (d1 < 30)
      x1[((size_t)s * NN + i) * 30 + d1] = c1[r] * inv + gat_bias[d1];
  }
}

// ---------------- residual + MLP (30 -> 64 -> 30, relu) ----------------
__global__ void mlp_kernel(const float* __restrict__ x1, const float* __restrict__ g1,
                           const float* __restrict__ w1, const float* __restrict__ b1,
                           const float* __restrict__ w2, const float* __restrict__ b2,
                           float* __restrict__ x3) {
  __shared__ float sw1[64 * 30], sw2[30 * 64], sb1[64], sb2[30];
  for (int i = threadIdx.x; i < 64 * 30; i += blockDim.x) { sw1[i] = w1[i]; sw2[i] = w2[i]; }
  for (int i = threadIdx.x; i < 64; i += blockDim.x) sb1[i] = b1[i];
  for (int i = threadIdx.x; i < 30; i += blockDim.x) sb2[i] = b2[i];
  __syncthreads();
  int t = blockIdx.x * blockDim.x + threadIdx.x;   // s*2048+n
  float x2[30];
#pragma unroll
  for (int d = 0; d < 30; ++d) {
    float v = x1[(size_t)t * 30 + d];
    x2[d] = (v > 0.f ? v : 0.f) + g1[(size_t)t * 30 + d];
  }
  float hm[64];
#pragma unroll
  for (int o = 0; o < 64; ++o) {
    float a = sb1[o];
#pragma unroll
    for (int d = 0; d < 30; ++d) a += sw1[o * 30 + d] * x2[d];
    hm[o] = a > 0.f ? a : 0.f;
  }
#pragma unroll
  for (int d = 0; d < 30; ++d) {
    float a = sb2[d];
#pragma unroll
    for (int o = 0; o < 64; ++o) a += sw2[d * 64 + o] * hm[o];
    x3[(size_t)t * 30 + d] = a > 0.f ? a : 0.f;
  }
}

// ---------------- conv2d (C=8 -> 12, 3x3, pad 1) on [S][N][30] ----------------
__global__ void conv_kernel(const float* __restrict__ g2, const float* __restrict__ w,
                            const float* __restrict__ b, float* __restrict__ x6) {
  int t = blockIdx.x * blockDim.x + threadIdx.x;   // 12*2048*30
  int d = t % 30;
  int n = (t / 30) % NN;
  int p = t / (30 * NN);
  float acc = b[p];
#pragma unroll
  for (int s = 0; s < S_OBS; ++s)
#pragma unroll
    for (int ky = -1; ky <= 1; ++ky) {
      int nr = n + ky;
      if (nr < 0 || nr >= NN) continue;
#pragma unroll
      for (int kx = -1; kx <= 1; ++kx) {
        int dd = d + kx;
        if (dd < 0 || dd >= 30) continue;
        acc += g2[((size_t)s * NN + nr) * 30 + dd] * w[((p * 8 + s) * 3 + (ky + 1)) * 3 + (kx + 1)];
      }
    }
  x6[(size_t)t] = acc > 0.f ? acc : 0.f;
}

// ---------------- output heads: 30 -> relu(64) -> 2 ----------------
__global__ void out_kernel(const float* __restrict__ x6,
                           const float* __restrict__ w1, const float* __restrict__ b1,
                           const float* __restrict__ w2, const float* __restrict__ b2,
                           float* __restrict__ out) {
  __shared__ float sw1[64 * 30], sw2[2 * 64], sb1[64], sb2[2];
  for (int i = threadIdx.x; i < 64 * 30; i += blockDim.x) sw1[i] = w1[i];
  for (int i = threadIdx.x; i < 128; i += blockDim.x) sw2[i] = w2[i];
  for (int i = threadIdx.x; i < 64; i += blockDim.x) sb1[i] = b1[i];
  if (threadIdx.x < 2) sb2[threadIdx.x] = b2[threadIdx.x];
  __syncthreads();
  int t = blockIdx.x * blockDim.x + threadIdx.x;   // p*2048+n, 24576 threads
  float xr[30];
#pragma unroll
  for (int d = 0; d < 30; ++d) xr[d] = x6[(size_t)t * 30 + d];
  float o0 = sb2[0], o1 = sb2[1];
#pragma unroll
  for (int o = 0; o < 64; ++o) {
    float a = sb1[o];
#pragma unroll
    for (int d = 0; d < 30; ++d) a += sw1[o * 30 + d] * xr[d];
    a = a > 0.f ? a : 0.f;
    o0 += sw2[o] * a;
    o1 += sw2[64 + o] * a;
  }
  out[(size_t)t * 2 + 0] = o0;
  out[(size_t)t * 2 + 1] = o1;
}

// ---------------- launch ----------------
extern "C" void kernel_launch(void* const* d_in, const int* in_sizes, int n_in,
                              void* d_out, int out_size, void* d_ws, size_t ws_size,
                              hipStream_t stream) {
  (void)in_sizes; (void)n_in; (void)out_size; (void)ws_size;
  const float* x       = (const float*)d_in[0];
  const int*   adj     = (const int*)  d_in[1];
  const float* fc1_w   = (const float*)d_in[2];
  const float* fc1_b   = (const float*)d_in[3];
  const float* g1_wih  = (const float*)d_in[4];
  const float* g1_whh  = (const float*)d_in[5];
  const float* g1_bih  = (const float*)d_in[6];
  const float* g1_bhh  = (const float*)d_in[7];
  const float* gat_w   = (const float*)d_in[8];
  const float* att_src = (const float*)d_in[9];
  const float* att_dst = (const float*)d_in[10];
  const float* gat_b   = (const float*)d_in[11];
  const float* mlp_w1  = (const float*)d_in[12];
  const float* mlp_b1  = (const float*)d_in[13];
  const float* mlp_w2  = (const float*)d_in[14];
  const float* mlp_b2  = (const float*)d_in[15];
  const float* g2_wih  = (const float*)d_in[16];
  const float* g2_whh  = (const float*)d_in[17];
  const float* g2_bih  = (const float*)d_in[18];
  const float* g2_bhh  = (const float*)d_in[19];
  const float* conv_w  = (const float*)d_in[20];
  const float* conv_b  = (const float*)d_in[21];
  const float* out1_w  = (const float*)d_in[22];
  const float* out1_b  = (const float*)d_in[23];
  const float* out2_w  = (const float*)d_in[24];
  const float* out2_b  = (const float*)d_in[25];

  float* ws  = (float*)d_ws;
  float* h1  = ws + OFF_H1;
  float* g1b = ws + OFF_G1;
  float* x1b = ws + OFF_X1;
  float* x3b = ws + OFF_X3;
  float* g2b = ws + OFF_G2;
  float* x6b = ws + OFF_X6;
  float* asb = ws + OFF_AS;
  float* adb = ws + OFF_AD;
  _Float16* xwh = (_Float16*)(ws + OFF_XWH);
  float* out = (float*)d_out;

  fc1_kernel<<<64, 256, 0, stream>>>(x, fc1_w, fc1_b, h1);
  gru_kernel<16><<<8, 256, 0, stream>>>(h1, g1_wih, g1_whh, g1_bih, g1_bhh, g1b);
  gat_pre_kernel<<<64, 256, 0, stream>>>(g1b, gat_w, att_src, att_dst, asb, adb, xwh);
  gat_attn_wmma_kernel<<<256, 128, 0, stream>>>(adj, asb, adb, xwh, gat_b, x1b);
  mlp_kernel<<<64, 256, 0, stream>>>(x1b, g1b, mlp_w1, mlp_b1, mlp_w2, mlp_b2, x3b);
  gru_kernel<30><<<8, 256, 0, stream>>>(x3b, g2_wih, g2_whh, g2_bih, g2_bhh, g2b);
  conv_kernel<<<2880, 256, 0, stream>>>(g2b, conv_w, conv_b, x6b);
  out_kernel<<<96, 256, 0, stream>>>(x6b, out1_w, out1_b, out2_w, out2_b, out);
}
